// Quantizer_33036888441545
// MI455X (gfx1250) — matrix operand order repaired and checked by reading, not compiled, not run on hardware
//
#include <hip/hip_runtime.h>
#include <stdint.h>
#include <stddef.h>

typedef __attribute__((ext_vector_type(16))) __bf16 v16bf;
typedef __attribute__((ext_vector_type(8)))  float  v8f;

#define DDIM 128
#define TILE_N 64
#define LDS_STRIDE 136   // 128 + 8 bf16 pad -> 272B row stride, conflict-free ds_load_b128
#define BETA 0.25f

// Load 16 contiguous bf16 from LDS as two 16B chunks (B-matrix operand).
__device__ inline v16bf lds_ld16(const __bf16* p) {
  union { uint4 u[2]; v16bf v; } t;
  t.u[0] = *(const uint4*)(p);
  t.u[1] = *(const uint4*)(p + 8);
  return t.v;
}

// Async copy 64 contiguous bytes global -> LDS (4x b128). INST_OFFSET is added
// to BOTH the LDS address (VDST) and the global address (VADDR) per ISA 10.7,
// so one base pair covers the whole chunk. Tracked by ASYNCcnt.
__device__ inline void async_cp64(uint32_t lds_addr, uint64_t gaddr) {
  asm volatile(
      "global_load_async_to_lds_b128 %0, %1, off\n\t"
      "global_load_async_to_lds_b128 %0, %1, off offset:16\n\t"
      "global_load_async_to_lds_b128 %0, %1, off offset:32\n\t"
      "global_load_async_to_lds_b128 %0, %1, off offset:48"
      :: "v"(lds_addr), "v"(gaddr) : "memory");
}

__device__ inline void wait_async0() {
  asm volatile("s_wait_asynccnt 0x0" ::: "memory");
}

// ---------------------------------------------------------------------------
// Kernel 1: split codes into bf16 hi/lo and compute exact fp32 ||c||^2.
// One wave (32 lanes) per code row; 8 rows per 256-thread block.
// ---------------------------------------------------------------------------
__global__ void __launch_bounds__(256) vq_prep(
    const float* __restrict__ codes,
    __bf16* __restrict__ chi, __bf16* __restrict__ clo,
    float* __restrict__ cnorm, int C) {
  int wave = threadIdx.x >> 5;
  int lane = threadIdx.x & 31;
  int row  = blockIdx.x * 8 + wave;
  if (row >= C) return;
  const float* src = codes + (size_t)row * DDIM + lane * 4;
  float4 v = *(const float4*)src;
  float f[4] = {v.x, v.y, v.z, v.w};
  union { __bf16 b[4]; uint2 u; } ph, pl;
  float s = 0.f;
#pragma unroll
  for (int i = 0; i < 4; ++i) {
    __bf16 h = (__bf16)f[i];
    ph.b[i] = h;
    pl.b[i] = (__bf16)(f[i] - (float)h);
    s += f[i] * f[i];
  }
  *(uint2*)(chi + (size_t)row * DDIM + lane * 4) = ph.u;
  *(uint2*)(clo + (size_t)row * DDIM + lane * 4) = pl.u;
#pragma unroll
  for (int m = 1; m < 32; m <<= 1) s += __shfl_xor(s, m, 32);
  if (lane == 0) cnorm[row] = s;
}

// ---------------------------------------------------------------------------
// Kernel 2: main search. 256 threads = 8 waves; each wave owns 16 x-rows.
// Code tiles of 64 stream through double-buffered LDS via async-to-LDS DMA;
// 3x bf16 WMMA per K=32 chunk (hi*hi + hi*lo + lo*hi) emulates fp32 dots.
// ---------------------------------------------------------------------------
__global__ void __launch_bounds__(256) vq_main(
    const float* __restrict__ x, const float* __restrict__ codes,
    const __bf16* __restrict__ chi, const __bf16* __restrict__ clo,
    const float* __restrict__ cnorm,
    float* __restrict__ qout, float* __restrict__ idxout,
    float* __restrict__ bpart, int C) {
  __shared__ __bf16 sh_hi[2][TILE_N * LDS_STRIDE];
  __shared__ __bf16 sh_lo[2][TILE_N * LDS_STRIDE];
  __shared__ float  xn[128];
  __shared__ float  ls[8];

  const int tid  = threadIdx.x;
  const int wave = tid >> 5;
  const int lane = tid & 31;
  const int col  = lane & 15;
  const int half = lane >> 4;
  const int rowBase = blockIdx.x * 128 + wave * 16;
  const int myRow   = rowBase + col;

  const int r = tid >> 2;  // tile row 0..63
  const int q = tid & 3;   // 64B quarter of a 256B row

  // Kick off tile 0 DMA immediately so it overlaps the A-register setup.
  {
    uint64_t gh = (uint64_t)(uintptr_t)(chi + (size_t)r * DDIM + q * 32);
    uint64_t gl = (uint64_t)(uintptr_t)(clo + (size_t)r * DDIM + q * 32);
    uint32_t lh = (uint32_t)(uintptr_t)&sh_hi[0][r * LDS_STRIDE + q * 32];
    uint32_t ll = (uint32_t)(uintptr_t)&sh_lo[0][r * LDS_STRIDE + q * 32];
    async_cp64(lh, gh);
    async_cp64(ll, gl);
  }

  // --- Load A (16 rows x 128 K) into WMMA layout, split hi/lo, row norms ---
  // A k order used here (same as B): lanes 0-15 (row=lane) hold K {0..15} per 32-chunk,
  // lanes 16-31 (row=lane-16) hold K {16..31}.
  v16bf a_hi[4], a_lo[4];
  float sumsq = 0.f;
  const float* xr = x + (size_t)myRow * DDIM;
#pragma unroll
  for (int kc = 0; kc < 4; ++kc) {
    int kb = kc * 32 + half * 16;
    float4 f0 = *(const float4*)(xr + kb);
    float4 f1 = *(const float4*)(xr + kb + 4);
    float4 f2 = *(const float4*)(xr + kb + 8);
    float4 f3 = *(const float4*)(xr + kb + 12);
    float f[16] = {f0.x, f0.y, f0.z, f0.w, f1.x, f1.y, f1.z, f1.w,
                   f2.x, f2.y, f2.z, f2.w, f3.x, f3.y, f3.z, f3.w};
#pragma unroll
    for (int i = 0; i < 16; ++i) {
      __bf16 h = (__bf16)f[i];
      a_hi[kc][i] = h;
      a_lo[kc][i] = (__bf16)(f[i] - (float)h);
      sumsq += f[i] * f[i];
    }
  }
  sumsq += __shfl_xor(sumsq, 16, 32);  // lanes L and L+16 cover disjoint K halves
  if (half == 0) xn[wave * 16 + col] = sumsq;
  __syncthreads();
  float xns[8];
#pragma unroll
  for (int s = 0; s < 8; ++s) xns[s] = xn[wave * 16 + half * 8 + s];

  float best[8]; int bidx[8];
#pragma unroll
  for (int s = 0; s < 8; ++s) { best[s] = 3.4e38f; bidx[s] = 0; }

  const int nIter = C / TILE_N;
  for (int it = 0; it < nIter; ++it) {
    const int ct  = it * TILE_N;
    const int buf = it & 1;

    // Our async copies for this buffer are done + everyone else's too.
    wait_async0();
    __syncthreads();

    // Prefetch the next tile into the other buffer while we compute.
    if (it + 1 < nIter) {
      int ctn = ct + TILE_N;
      uint64_t gh = (uint64_t)(uintptr_t)(chi + (size_t)(ctn + r) * DDIM + q * 32);
      uint64_t gl = (uint64_t)(uintptr_t)(clo + (size_t)(ctn + r) * DDIM + q * 32);
      uint32_t lh = (uint32_t)(uintptr_t)&sh_hi[buf ^ 1][r * LDS_STRIDE + q * 32];
      uint32_t ll = (uint32_t)(uintptr_t)&sh_lo[buf ^ 1][r * LDS_STRIDE + q * 32];
      async_cp64(lh, gh);
      async_cp64(ll, gl);
    }

    const __bf16* th = &sh_hi[buf][0];
    const __bf16* tl = &sh_lo[buf][0];
#pragma unroll
    for (int n0 = 0; n0 < TILE_N; n0 += 16) {
      v8f acc = {};
      // B k order used here (same as A): lane n<16 = col n, K 0..15; lane n+16 = col n, K 16..31
      // -> each lane reads 16 contiguous bf16 of its code row.
#pragma unroll
      for (int kc = 0; kc < 4; ++kc) {
        v16bf bh = lds_ld16(th + (n0 + col) * LDS_STRIDE + kc * 32 + half * 16);
        v16bf bl = lds_ld16(tl + (n0 + col) * LDS_STRIDE + kc * 32 + half * 16);
        acc = __builtin_amdgcn_wmma_f32_16x16x32_bf16(false, a_hi[kc], false, bh,
                                                      (short)0, acc, false, false);
        acc = __builtin_amdgcn_wmma_f32_16x16x32_bf16(false, a_lo[kc], false, bh,
                                                      (short)0, acc, false, false);
        acc = __builtin_amdgcn_wmma_f32_16x16x32_bf16(false, a_hi[kc], false, bl,
                                                      (short)0, acc, false, false);
      }
      int ci = ct + n0 + col;
      float cn = cnorm[ci];
      // C/D layout: slot s -> row (s + 8*half), col = lane&15
#pragma unroll
      for (int s = 0; s < 8; ++s) {
        float d2 = xns[s] + cn - 2.f * acc[s];
        if (d2 < best[s]) { best[s] = d2; bidx[s] = ci; }
      }
    }
  }

  // Argmin across the 16 columns held by each half-wave (xor<16 stays in half);
  // tie-break to the smaller index (jnp.argmin semantics).
#pragma unroll
  for (int s = 0; s < 8; ++s) {
    float m = best[s]; int bi = bidx[s];
#pragma unroll
    for (int msk = 1; msk <= 8; msk <<= 1) {
      float om = __shfl_xor(m, msk, 32);
      int   oi = __shfl_xor(bi, msk, 32);
      if (om < m || (om == m && oi < bi)) { m = om; bi = oi; }
    }
    best[s] = m; bidx[s] = bi;
  }

  // Gather quantized rows in exact fp32, recompute exact ||x-q||^2, write idx.
  float lpart = 0.f;
#pragma unroll
  for (int s = 0; s < 8; ++s) {
    int row = rowBase + half * 8 + s;
    int ci  = bidx[s];
    const float* crow = codes + (size_t)ci * DDIM + col * 8;
    const float* xrow = x + (size_t)row * DDIM + col * 8;
    float* qrow = qout + (size_t)row * DDIM + col * 8;
    float4 c0 = *(const float4*)crow;
    float4 c1 = *(const float4*)(crow + 4);
    float4 x0 = *(const float4*)xrow;
    float4 x1 = *(const float4*)(xrow + 4);
    *(float4*)qrow       = c0;
    *(float4*)(qrow + 4) = c1;
    float d;
    d = x0.x - c0.x; lpart += d * d;
    d = x0.y - c0.y; lpart += d * d;
    d = x0.z - c0.z; lpart += d * d;
    d = x0.w - c0.w; lpart += d * d;
    d = x1.x - c1.x; lpart += d * d;
    d = x1.y - c1.y; lpart += d * d;
    d = x1.z - c1.z; lpart += d * d;
    d = x1.w - c1.w; lpart += d * d;
    if (col == 0) idxout[row] = (float)ci;
  }
#pragma unroll
  for (int msk = 1; msk < 32; msk <<= 1) lpart += __shfl_xor(lpart, msk, 32);
  if (lane == 0) ls[wave] = lpart;
  __syncthreads();
  if (tid == 0) {
    float s = 0.f;
#pragma unroll
    for (int w = 0; w < 8; ++w) s += ls[w];
    bpart[blockIdx.x] = s;  // fixed-order, deterministic
  }
}

// ---------------------------------------------------------------------------
// Kernel 3: deterministic serial finalize of the loss.
// ---------------------------------------------------------------------------
__global__ void vq_fin(const float* __restrict__ bpart, int nb, int B,
                       float* __restrict__ loss) {
  if (threadIdx.x == 0 && blockIdx.x == 0) {
    float s = 0.f;
    for (int i = 0; i < nb; ++i) s += bpart[i];
    *loss = s * (1.f + BETA) / (float)B;
  }
}

extern "C" void kernel_launch(void* const* d_in, const int* in_sizes, int n_in,
                              void* d_out, int out_size, void* d_ws, size_t ws_size,
                              hipStream_t stream) {
  const float* x     = (const float*)d_in[0];
  const float* codes = (const float*)d_in[1];
  const int B = in_sizes[0] / DDIM;   // 65536
  const int C = in_sizes[1] / DDIM;   // 4096

  float* qout    = (float*)d_out;                 // [B, 128] fp32
  float* idxout  = qout + (size_t)B * DDIM;       // [B] indices (as float)
  float* lossout = idxout + B;                    // [1]

  char* ws = (char*)d_ws;
  __bf16* chi  = (__bf16*)ws;                                  // C*128 bf16 (1 MB)
  __bf16* clo  = (__bf16*)(ws + (size_t)C * DDIM * 2);         // C*128 bf16 (1 MB)
  float* cnorm = (float*)(ws + (size_t)C * DDIM * 4);          // C floats
  float* bpart = (float*)(ws + (size_t)C * DDIM * 4 + (size_t)C * 4);  // B/128 floats

  const int nb = B / 128;
  vq_prep<<<(C + 7) / 8, 256, 0, stream>>>(codes, chi, clo, cnorm, C);
  vq_main<<<nb, 256, 0, stream>>>(x, codes, chi, clo, cnorm, qout, idxout, bpart, C);
  vq_fin<<<1, 32, 0, stream>>>(bpart, nb, B, lossout);
}
